// MultiHeadAttention_23862838297407
// MI455X (gfx1250) — compile-verified, matrix operand order checked by reading, not run on hardware
//
#include <hip/hip_runtime.h>

typedef __attribute__((ext_vector_type(16))) __bf16 v16bf;
typedef __attribute__((ext_vector_type(8)))  float  v8f;
typedef __attribute__((ext_vector_type(4)))  unsigned su4;
typedef __attribute__((ext_vector_type(8)))  unsigned su8;

#define BB 4
#define SS 2048
#define HH 16
#define DH 64
#define IN_DIM 1024
#define N3 3072
#define OUT_DIM 1024
#define MROWS (BB*SS)

typedef unsigned short u16;

union Frag { v16bf v; uint4 q[2]; };

static __device__ __forceinline__ v16bf load_frag(const u16* p0, const u16* p1) {
  Frag f;
  f.q[0] = *reinterpret_cast<const uint4*>(p0);
  f.q[1] = *reinterpret_cast<const uint4*>(p1);
  return f.v;
}

// fragment from an LDS tile row of 32 bf16: chunks at +0 and +16 elements
static __device__ __forceinline__ v16bf lds_frag(const u16* base) {
  Frag f;
  f.q[0] = *reinterpret_cast<const uint4*>(base);
  f.q[1] = *reinterpret_cast<const uint4*>(base + 16);
  return f.v;
}

static __device__ __forceinline__ v8f wmma_bf16(v16bf a, v16bf b, v8f c) {
  return __builtin_amdgcn_wmma_f32_16x16x32_bf16(false, a, false, b, (short)0, c, false, false);
}

static __device__ __forceinline__ u16 f2bf(float f) {
  unsigned u = __float_as_uint(f);
  unsigned r = u + 0x7FFFu + ((u >> 16) & 1u);   // round-to-nearest-even
  return (u16)(r >> 16);
}

// ---------------------------------------------------------------- TDM 2-D tile load
// D# per CDNA5 ISA §8.3/8.4: group0 {count=1, lds_addr, global_addr, type=2},
// group1 {data_size=2B, tensor_dim0/1, tile_dim0/1, tensor_dim0_stride}.
static __device__ __forceinline__ void tdm_load_tile_2d(const u16* gsrc,
                                                        unsigned lds_byte,
                                                        unsigned rows_total,
                                                        unsigned row_stride_elems,
                                                        unsigned tile_k,
                                                        unsigned tile_rows) {
  unsigned long long ga = (unsigned long long)(uintptr_t)gsrc;
  unsigned ga_lo = __builtin_amdgcn_readfirstlane((unsigned)ga);
  unsigned ga_hi = __builtin_amdgcn_readfirstlane((unsigned)(ga >> 32));
  su4 g0;
  g0[0] = 1u;                                   // count=1, user descriptor
  g0[1] = __builtin_amdgcn_readfirstlane(lds_byte);
  g0[2] = ga_lo;
  g0[3] = ga_hi | (2u << 30);                   // type=2 ("image")
  su8 g1;
  g1[0] = 1u << 16;                             // workgroup_mask=0, data_size=1 (2 bytes)
  g1[1] = (row_stride_elems & 0xFFFFu) << 16;   // tensor_dim0[15:0] @ bits 63:48
  g1[2] = (row_stride_elems >> 16) | ((rows_total & 0xFFFFu) << 16); // dim0 hi / dim1 lo
  g1[3] = (rows_total >> 16) | (tile_k << 16);  // dim1 hi / tile_dim0
  g1[4] = tile_rows & 0xFFFFu;                  // tile_dim1 (tile_dim2=0)
  g1[5] = row_stride_elems;                     // tensor_dim0_stride[31:0]
  g1[6] = 0u;                                   // stride hi / dim1_stride lo
  g1[7] = 0u;
  asm volatile("tensor_load_to_lds %0, %1" :: "s"(g0), "s"(g1) : "memory");
}

// ---------------------------------------------------------------- convert
__global__ void mha_cvt_bf16(const float* __restrict__ src, u16* __restrict__ dst, int n) {
  int i = blockIdx.x * blockDim.x + threadIdx.x;
  int stride = gridDim.x * blockDim.x;
  for (; i < n; i += stride) dst[i] = f2bf(src[i]);
}

// ---------------------------------------------------------------- QKV GEMM
// Block computes 128M x 128N; TDM double-buffers 128x32 A/B bf16 tiles in LDS.
// Wave (4Mx2N grid) computes 32x64 via 8 WMMAs per K-step.
__global__ void __launch_bounds__(256) mha_qkv_gemm(const u16* __restrict__ Xb,
                                                    const u16* __restrict__ Wb,
                                                    u16* __restrict__ Qb,
                                                    u16* __restrict__ Kb,
                                                    u16* __restrict__ Vtb) {
  __shared__ u16 Atile[2][128 * 32];
  __shared__ u16 Btile[2][128 * 32];
  const int wid  = threadIdx.x >> 5;
  const int lane = threadIdx.x & 31;
  const int half = lane >> 4;
  const int ln   = lane & 15;
  const int wm   = wid & 3;          // M quarter (32 rows)
  const int wn   = wid >> 2;         // N half (64 cols)
  const int mblk = blockIdx.x, nblk = blockIdx.y;

  const u16* Ag = Xb + (size_t)(mblk * 128) * IN_DIM;
  const u16* Bg = Wb + (size_t)(nblk * 128) * IN_DIM;
  const unsigned ldsA[2] = { (unsigned)(uintptr_t)&Atile[0][0], (unsigned)(uintptr_t)&Atile[1][0] };
  const unsigned ldsB[2] = { (unsigned)(uintptr_t)&Btile[0][0], (unsigned)(uintptr_t)&Btile[1][0] };

  if (wid == 0) {
    tdm_load_tile_2d(Ag, ldsA[0], MROWS, IN_DIM, 32, 128);
    tdm_load_tile_2d(Bg, ldsB[0], N3,    IN_DIM, 32, 128);
  }

  v8f acc[2][4] = {};
  const int KSTEPS = IN_DIM / 32;
  for (int i = 0; i < KSTEPS; ++i) {
    if (wid == 0) {
      if (i + 1 < KSTEPS) {
        const int kk = (i + 1) * 32;
        tdm_load_tile_2d(Ag + kk, ldsA[(i + 1) & 1], MROWS, IN_DIM, 32, 128);
        tdm_load_tile_2d(Bg + kk, ldsB[(i + 1) & 1], N3,    IN_DIM, 32, 128);
        __builtin_amdgcn_s_wait_tensorcnt(2);   // current pair done, next in flight
      } else {
        __builtin_amdgcn_s_wait_tensorcnt(0);
      }
    }
    __syncthreads();
    const u16* At = Atile[i & 1];
    const u16* Bt = Btile[i & 1];
    const int c0 = half * 8;
    v16bf a0 = lds_frag(&At[(wm * 32 + ln) * 32 + c0]);
    v16bf a1 = lds_frag(&At[(wm * 32 + 16 + ln) * 32 + c0]);
#pragma unroll
    for (int nt = 0; nt < 4; ++nt) {
      v16bf b = lds_frag(&Bt[(wn * 64 + nt * 16 + ln) * 32 + c0]);
      acc[0][nt] = wmma_bf16(a0, b, acc[0][nt]);
      acc[1][nt] = wmma_bf16(a1, b, acc[1][nt]);
    }
    __syncthreads();
  }

#pragma unroll
  for (int mt = 0; mt < 2; ++mt)
#pragma unroll
    for (int nt = 0; nt < 4; ++nt) {
      const int gc  = nblk * 128 + wn * 64 + nt * 16 + ln;
      const int hh  = gc / 192;
      const int rem = gc - hh * 192;
      const int d   = rem / 3;
      const int wh  = rem - d * 3;              // 0=k 1=q 2=v
#pragma unroll
      for (int r = 0; r < 8; ++r) {
        const int gm = mblk * 128 + wm * 32 + mt * 16 + half * 8 + r;
        const int bb = gm >> 11;
        const int s  = gm & (SS - 1);
        const u16 val = f2bf(acc[mt][nt][r]);
        if (wh == 2) {
          Vtb[(((size_t)bb * HH + hh) * DH + d) * SS + s] = val;   // transposed V
        } else {
          u16* p = (wh == 0) ? Kb : Qb;
          p[(((size_t)bb * HH + hh) * SS + s) * DH + d] = val;
        }
      }
    }
}

// ---------------------------------------------------------------- flash attention
// One wave per 16 query rows; causal online softmax; key steps of 32.
__global__ void __launch_bounds__(256) mha_flash(const u16* __restrict__ Qb,
                                                 const u16* __restrict__ Kb,
                                                 const u16* __restrict__ Vtb,
                                                 u16* __restrict__ Ctx) {
  __shared__ u16 plds[8][16 * 32];   // per-wave P staging (1 KB each)
  const int wid  = threadIdx.x >> 5;
  const int lane = threadIdx.x & 31;
  const int half = lane >> 4;
  const int ln   = lane & 15;
  const int qtile = blockIdx.x * 8 + wid;
  const int q0 = qtile * 16;
  const int h = blockIdx.y, b = blockIdx.z;
  const int bh = b * HH + h;

  const u16* Qp = Qb  + (size_t)bh * SS * DH;
  const u16* Kp = Kb  + (size_t)bh * SS * DH;
  const u16* Vp = Vtb + (size_t)bh * DH * SS;

  const int c0 = half * 8;
  const u16* qrow = Qp + (size_t)(q0 + ln) * DH;
  v16bf a0 = load_frag(qrow + c0,      qrow + c0 + 16);   // d 0..31
  v16bf a1 = load_frag(qrow + 32 + c0, qrow + 48 + c0);   // d 32..63

  v8f o[4] = {};
  float mrow[8], lrow[8];
#pragma unroll
  for (int r = 0; r < 8; ++r) { mrow[r] = -1e30f; lrow[r] = 0.f; }

  for (int j = 0; j < q0 + 16; j += 32) {
    const u16* krow0 = Kp + (size_t)(j + ln) * DH;
    const u16* krow1 = krow0 + 16 * DH;
    v16bf kb00 = load_frag(krow0 + c0,      krow0 + c0 + 16);
    v16bf kb01 = load_frag(krow0 + 32 + c0, krow0 + 48 + c0);
    v16bf kb10 = load_frag(krow1 + c0,      krow1 + c0 + 16);
    v16bf kb11 = load_frag(krow1 + 32 + c0, krow1 + 48 + c0);

    v8f s0 = {}, s1 = {};
    s0 = wmma_bf16(a0, kb00, s0);  s0 = wmma_bf16(a1, kb01, s0);
    s1 = wmma_bf16(a0, kb10, s1);  s1 = wmma_bf16(a1, kb11, s1);

    const int col0 = j + ln, col1 = col0 + 16;
    const int rb = q0 + half * 8;
#pragma unroll
    for (int r = 0; r < 8; ++r) {
      float x0 = s0[r] * 0.125f; if (col0 > rb + r) x0 = -1e30f;  // D^-0.5 + causal
      float x1 = s1[r] * 0.125f; if (col1 > rb + r) x1 = -1e30f;
      float mx = fmaxf(x0, x1);
      mx = fmaxf(mx, __shfl_xor(mx, 1));
      mx = fmaxf(mx, __shfl_xor(mx, 2));
      mx = fmaxf(mx, __shfl_xor(mx, 4));
      mx = fmaxf(mx, __shfl_xor(mx, 8));
      const float mnew = fmaxf(mrow[r], mx);
      const float scl  = __expf(mrow[r] - mnew);
      const float p0   = __expf(x0 - mnew);
      const float p1   = __expf(x1 - mnew);
      float rs = p0 + p1;
      rs += __shfl_xor(rs, 1); rs += __shfl_xor(rs, 2);
      rs += __shfl_xor(rs, 4); rs += __shfl_xor(rs, 8);
      lrow[r] = lrow[r] * scl + rs;
      mrow[r] = mnew;
      o[0][r] *= scl; o[1][r] *= scl; o[2][r] *= scl; o[3][r] *= scl;
      const int rloc = half * 8 + r;
      plds[wid][rloc * 32 + ln]      = f2bf(p0);
      plds[wid][rloc * 32 + ln + 16] = f2bf(p1);
    }
    asm volatile("s_wait_dscnt 0" ::: "memory");

    Frag pf;   // re-read P in A-fragment order: lane=row ln, chunks at k=c0 / c0+16
    pf.q[0] = *reinterpret_cast<const uint4*>(&plds[wid][ln * 32 + c0]);
    pf.q[1] = *reinterpret_cast<const uint4*>(&plds[wid][ln * 32 + 16 + c0]);

#pragma unroll
    for (int t = 0; t < 4; ++t) {
      const u16* vrow = Vp + (size_t)(t * 16 + ln) * SS + j;   // Vt row d, contiguous s
      v16bf vb = load_frag(vrow + c0, vrow + 16 + c0);
      o[t] = wmma_bf16(pf.v, vb, o[t]);
    }
  }

#pragma unroll
  for (int r = 0; r < 8; ++r) {
    const float inv = 1.0f / lrow[r];
    const int gq = q0 + half * 8 + r;
    const size_t base = ((size_t)b * SS + gq) * OUT_DIM + h * DH + ln;
#pragma unroll
    for (int t = 0; t < 4; ++t)
      Ctx[base + t * 16] = f2bf(o[t][r] * inv);
  }
}

// ---------------------------------------------------------------- output projection
__global__ void __launch_bounds__(256) mha_proj_gemm(const u16* __restrict__ Ctx,
                                                     const u16* __restrict__ Wb,
                                                     const float* __restrict__ bias,
                                                     float* __restrict__ out) {
  __shared__ u16 Atile[2][128 * 32];
  __shared__ u16 Btile[2][128 * 32];
  const int wid  = threadIdx.x >> 5;
  const int lane = threadIdx.x & 31;
  const int half = lane >> 4;
  const int ln   = lane & 15;
  const int wm   = wid & 3;
  const int wn   = wid >> 2;
  const int mblk = blockIdx.x, nblk = blockIdx.y;

  const u16* Ag = Ctx + (size_t)(mblk * 128) * OUT_DIM;
  const u16* Bg = Wb  + (size_t)(nblk * 128) * OUT_DIM;
  const unsigned ldsA[2] = { (unsigned)(uintptr_t)&Atile[0][0], (unsigned)(uintptr_t)&Atile[1][0] };
  const unsigned ldsB[2] = { (unsigned)(uintptr_t)&Btile[0][0], (unsigned)(uintptr_t)&Btile[1][0] };

  if (wid == 0) {
    tdm_load_tile_2d(Ag, ldsA[0], MROWS,   OUT_DIM, 32, 128);
    tdm_load_tile_2d(Bg, ldsB[0], OUT_DIM, OUT_DIM, 32, 128);
  }

  v8f acc[2][4] = {};
  const int KSTEPS = OUT_DIM / 32;
  for (int i = 0; i < KSTEPS; ++i) {
    if (wid == 0) {
      if (i + 1 < KSTEPS) {
        const int kk = (i + 1) * 32;
        tdm_load_tile_2d(Ag + kk, ldsA[(i + 1) & 1], MROWS,   OUT_DIM, 32, 128);
        tdm_load_tile_2d(Bg + kk, ldsB[(i + 1) & 1], OUT_DIM, OUT_DIM, 32, 128);
        __builtin_amdgcn_s_wait_tensorcnt(2);
      } else {
        __builtin_amdgcn_s_wait_tensorcnt(0);
      }
    }
    __syncthreads();
    const u16* At = Atile[i & 1];
    const u16* Bt = Btile[i & 1];
    const int c0 = half * 8;
    v16bf a0 = lds_frag(&At[(wm * 32 + ln) * 32 + c0]);
    v16bf a1 = lds_frag(&At[(wm * 32 + 16 + ln) * 32 + c0]);
#pragma unroll
    for (int nt = 0; nt < 4; ++nt) {
      v16bf b = lds_frag(&Bt[(wn * 64 + nt * 16 + ln) * 32 + c0]);
      acc[0][nt] = wmma_bf16(a0, b, acc[0][nt]);
      acc[1][nt] = wmma_bf16(a1, b, acc[1][nt]);
    }
    __syncthreads();
  }

#pragma unroll
  for (int mt = 0; mt < 2; ++mt)
#pragma unroll
    for (int nt = 0; nt < 4; ++nt) {
      const int gc = nblk * 128 + wn * 64 + nt * 16 + ln;
      const float bv = bias[gc];
#pragma unroll
      for (int r = 0; r < 8; ++r) {
        const int gm = mblk * 128 + wm * 32 + mt * 16 + half * 8 + r;
        out[(size_t)gm * OUT_DIM + gc] = acc[mt][nt][r] + bv;
      }
    }
}

// ---------------------------------------------------------------- launch
extern "C" void kernel_launch(void* const* d_in, const int* in_sizes, int n_in,
                              void* d_out, int out_size, void* d_ws, size_t ws_size,
                              hipStream_t stream) {
  const float* x      = (const float*)d_in[0];
  const float* w_kqv  = (const float*)d_in[1];
  const float* w_proj = (const float*)d_in[2];
  const float* b_proj = (const float*)d_in[3];
  float* out = (float*)d_out;

  char* ws = (char*)d_ws;
  u16* Xb   = (u16*)ws;  ws += (size_t)MROWS * IN_DIM * 2;     // 16 MB
  u16* Wkb  = (u16*)ws;  ws += (size_t)N3 * IN_DIM * 2;        // 6 MB
  u16* Wpb  = (u16*)ws;  ws += (size_t)OUT_DIM * OUT_DIM * 2;  // 2 MB
  u16* Qb   = (u16*)ws;  ws += (size_t)BB * HH * SS * DH * 2;  // 16 MB
  u16* Kb   = (u16*)ws;  ws += (size_t)BB * HH * SS * DH * 2;  // 16 MB
  u16* Vtb  = (u16*)ws;  ws += (size_t)BB * HH * DH * SS * 2;  // 16 MB
  u16* Ctx  = (u16*)ws;                                        // 16 MB

  const int nx = MROWS * IN_DIM, nk = N3 * IN_DIM, np = OUT_DIM * OUT_DIM;
  mha_cvt_bf16<<<(nx + 4095) / 4096, 256, 0, stream>>>(x, Xb, nx);
  mha_cvt_bf16<<<(nk + 4095) / 4096, 256, 0, stream>>>(w_kqv, Wkb, nk);
  mha_cvt_bf16<<<(np + 4095) / 4096, 256, 0, stream>>>(w_proj, Wpb, np);

  mha_qkv_gemm<<<dim3(MROWS / 128, N3 / 128), 256, 0, stream>>>(Xb, Wkb, Qb, Kb, Vtb);

  mha_flash<<<dim3(SS / 128, HH, BB), 256, 0, stream>>>(Qb, Kb, Vtb, Ctx);

  mha_proj_gemm<<<dim3(MROWS / 128, OUT_DIM / 128), 256, 0, stream>>>(Ctx, Wpb, b_proj, out);
}
